// MultiHeadAttention_35931696398381
// MI455X (gfx1250) — matrix operand order repaired and checked by reading, not compiled, not run on hardware
//
#include <hip/hip_runtime.h>

// MHA: B=2, S=2048, E=1024, H=16, D=64, fp32 in/out, bf16 WMMA internally.
#define S_LEN  2048
#define E_DIM  1024
#define NHEAD  16
#define D_HEAD 64
#define M_ROWS 4096  // B*S

typedef __bf16 v16bf __attribute__((ext_vector_type(16)));
typedef __bf16 bf4   __attribute__((ext_vector_type(4)));
typedef float  v8f   __attribute__((ext_vector_type(8)));

union FragU { v16bf v; uint4 q[2]; };

// gfx1250 async global->LDS copy path (ASYNCcnt). Guarded: falls back to
// register staging if the builtins are absent on this toolchain.
// Probe-learned signature: (v4i addrspace(1)*, v4i addrspace(3)*, imm, imm).
#if defined(__AMDGCN__) &&                                              \
    __has_builtin(__builtin_amdgcn_global_load_async_to_lds_b128) &&    \
    __has_builtin(__builtin_amdgcn_s_wait_asynccnt)
#define USE_ASYNC_LDS 1
typedef int async_v4i __attribute__((vector_size(16)));
#define AS_G1(p) ((__attribute__((address_space(1))) async_v4i*)(p))
#define AS_L3(p) ((__attribute__((address_space(3))) async_v4i*)(p))
#else
#define USE_ASYNC_LDS 0
#endif

static __device__ inline v8f wmma_bf16(v16bf a, v16bf b, v8f c) {
  // 8 args: (neg_a, A, neg_b, B, c_mod, C, reuse_a, reuse_b)
  return __builtin_amdgcn_wmma_f32_16x16x32_bf16(false, a, false, b,
                                                 (short)0, c, false, false);
}

// A-operand fragment (16x32, M x K): lanes 0-15 hold K 0..15,
// lanes 16-31 hold K 16..31, row = lane%16. rowBase points to K=0.
static __device__ inline v16bf frag_rowA(const __bf16* rowBase, int khalf) {
  FragU f;
  f.q[0] = *(const uint4*)(rowBase + khalf * 16);
  f.q[1] = *(const uint4*)(rowBase + khalf * 16 + 8);
  return f.v;
}

// B-operand fragment (32x16, K x N): lane holds one column; lanes 0-15 hold
// K 0..15, lanes 16-31 hold K 16..31. p points at the lane's 16-K chunk.
static __device__ inline v16bf frag_colB(const __bf16* p) {
  FragU f;
  f.q[0] = *(const uint4*)(p);
  f.q[1] = *(const uint4*)(p + 8);
  return f.v;
}

// ---- GEMM tile staging helpers (128 rows x 32 K per tile) ----------------
template <bool SRC_BF16>
static __device__ inline void tile_load_regs(const void* Ap, int K, int base,
                                             int kk, int tid, float4 gf[4],
                                             uint4 gb[2]) {
  if constexpr (SRC_BF16) {
    const __bf16* A = (const __bf16*)Ap;
#pragma unroll
    for (int i = 0; i < 2; ++i) {
      int s = tid + i * 256;
      int r = s >> 2, c8 = s & 3;
      gb[i] = *(const uint4*)&A[(size_t)(base + r) * K + kk + c8 * 8];
    }
  } else {
    const float* A = (const float*)Ap;
#pragma unroll
    for (int i = 0; i < 4; ++i) {
      int s = tid + i * 256;
      int r = s >> 3, c4 = s & 7;
      gf[i] = *(const float4*)&A[(size_t)(base + r) * K + kk + c4 * 4];
    }
  }
}

template <bool SRC_BF16>
static __device__ inline void tile_store_lds(__bf16* dst, int tid,
                                             const float4 gf[4],
                                             const uint4 gb[2]) {
  if constexpr (SRC_BF16) {
#pragma unroll
    for (int i = 0; i < 2; ++i) {
      int s = tid + i * 256;
      int r = s >> 2, c8 = s & 3;
      *(uint4*)&dst[r * 40 + c8 * 8] = gb[i];
    }
  } else {
#pragma unroll
    for (int i = 0; i < 4; ++i) {
      int s = tid + i * 256;
      int r = s >> 3, c4 = s & 7;
      float4 f = gf[i];
      bf4 hh = { (__bf16)f.x, (__bf16)f.y, (__bf16)f.z, (__bf16)f.w };
      *(bf4*)&dst[r * 40 + c4 * 4] = hh;
    }
  }
}

// C = A @ W^T + bias.  A: [M,K] (fp32 or bf16), W: [N,K] fp32 row-major,
// C: [M,N] (bf16 or fp32). Block = 128x128 tile, 256 threads = 8 waves,
// each wave computes a 32x64 region as 2x4 WMMA 16x16 tiles. K-step 32,
// double-buffered LDS: one barrier per step, next tile's global loads are
// issued before the current step's WMMAs.
template <bool A_BF16, bool OUT_BF16>
__global__ __launch_bounds__(256) void gemm_bias_wmma(
    const void* __restrict__ Ap, const float* __restrict__ W,
    const float* __restrict__ bias, void* __restrict__ Cp,
    int M, int N, int K) {
  const int tid   = threadIdx.x;
  const int wave  = tid >> 5;
  const int lane  = tid & 31;
  const int lmod  = lane & 15;
  const int khalf = lane >> 4;
  const int bm    = blockIdx.y * 128;
  const int bn    = blockIdx.x * 128;
  const int wrow  = (wave & 3) * 32;
  const int wcol  = (wave >> 2) * 64;

  __shared__ __bf16 As[2][128 * 40];
  __shared__ __bf16 Bs[2][128 * 40];

  float4 gaA[4]; uint4 gbA[2];
  float4 gaW[4]; uint4 gbW[2];

  // prologue: stage K-slab 0 into buffer 0
  tile_load_regs<A_BF16>(Ap, K, bm, 0, tid, gaA, gbA);
  tile_load_regs<false>(W, K, bn, 0, tid, gaW, gbW);
  tile_store_lds<A_BF16>(As[0], tid, gaA, gbA);
  tile_store_lds<false>(Bs[0], tid, gaW, gbW);

  v8f acc[2][4] = {};

  int cur = 0;
#pragma unroll 1
  for (int k0 = 0; k0 < K; k0 += 32, cur ^= 1) {
    __syncthreads();  // buf[cur] staged; previous reads of buf[cur^1] done
    const bool hasNext = (k0 + 32) < K;
    if (hasNext) {  // issue next tile's global loads now (overlap with WMMA)
      tile_load_regs<A_BF16>(Ap, K, bm, k0 + 32, tid, gaA, gbA);
      tile_load_regs<false>(W, K, bn, k0 + 32, tid, gaW, gbW);
      if (k0 + 64 < K)  // L2 prefetch two tiles ahead -> global_prefetch_b8
        __builtin_prefetch(
            &W[(size_t)(bn + (tid >> 3)) * K + k0 + 64 + (tid & 7) * 4], 0, 1);
    }

    v16bf af[2], bfv[4];
#pragma unroll
    for (int mi = 0; mi < 2; ++mi)
      af[mi] = frag_rowA(&As[cur][(wrow + mi * 16 + lmod) * 40], khalf);
#pragma unroll
    for (int ni = 0; ni < 4; ++ni)
      bfv[ni] = frag_colB(&Bs[cur][(wcol + ni * 16 + lmod) * 40 + khalf * 16]);
#pragma unroll
    for (int mi = 0; mi < 2; ++mi)
#pragma unroll
      for (int ni = 0; ni < 4; ++ni)
        acc[mi][ni] = wmma_bf16(af[mi], bfv[ni], acc[mi][ni]);

    if (hasNext) {  // convert + publish next tile after the WMMAs
      tile_store_lds<A_BF16>(As[cur ^ 1], tid, gaA, gbA);
      tile_store_lds<false>(Bs[cur ^ 1], tid, gaW, gbW);
    }
  }

  float bv[4];
#pragma unroll
  for (int ni = 0; ni < 4; ++ni) bv[ni] = bias[bn + wcol + ni * 16 + lmod];

#pragma unroll
  for (int mi = 0; mi < 2; ++mi)
#pragma unroll
    for (int ni = 0; ni < 4; ++ni)
#pragma unroll
      for (int r = 0; r < 8; ++r) {
        // C layout: lanes 0-15 -> M = r, lanes 16-31 -> M = r+8; N = lane%16
        int row = bm + wrow + mi * 16 + r + 8 * khalf;
        int col = bn + wcol + ni * 16 + lmod;
        float v = acc[mi][ni][r] + bv[ni];
        if constexpr (OUT_BF16)
          ((__bf16*)Cp)[(size_t)row * N + col] = (__bf16)v;
        else
          ((float*)Cp)[(size_t)row * N + col] = v;
      }
}

// Flash attention: one (64-q-rows, head, batch) tile per block; 4 waves,
// each wave owns 16 q rows. KV streamed in 32-wide double-buffered tiles.
// K tile staged with gfx1250 async global->LDS (ASYNCcnt) when available.
__global__ __launch_bounds__(128) void flash_attn_wmma(
    const __bf16* __restrict__ Qg, const __bf16* __restrict__ Kg,
    const __bf16* __restrict__ Vg, const int* __restrict__ mask,
    __bf16* __restrict__ Ctx) {
  const int tid   = threadIdx.x;
  const int wave  = tid >> 5;
  const int lane  = tid & 31;
  const int lmod  = lane & 15;
  const int khalf = lane >> 4;
  const int qt = blockIdx.x;
  const int h  = blockIdx.y;
  const int b  = blockIdx.z;

  __shared__ __bf16 Ks[2][32 * 72];      // 32 kv rows x 64 d (stride 72)
  __shared__ __bf16 Vt[2][64 * 40];      // transposed: 64 d rows x 32 kv
  __shared__ __bf16 Pb[4 * 16 * 40];     // per-wave P re-layout buffer

  const int qBase = qt * 64 + wave * 16;
  const __bf16* Qrow =
      Qg + ((size_t)b * S_LEN + qBase + lmod) * E_DIM + h * D_HEAD;
  v16bf qf[2];
#pragma unroll
  for (int kq = 0; kq < 2; ++kq) qf[kq] = frag_rowA(Qrow + kq * 32, khalf);

  const __bf16* Kbase = Kg + (size_t)b * S_LEN * E_DIM + h * D_HEAD;
  const __bf16* Vbase = Vg + (size_t)b * S_LEN * E_DIM + h * D_HEAD;
  __bf16* Pw = &Pb[wave * 16 * 40];

  auto stageK = [&](int buf, int kt) {  // 32x64 bf16 copy, 16B per lane x2
#pragma unroll
    for (int i = 0; i < 2; ++i) {
      int s = tid + i * 128;
      int r = s >> 3, c8 = s & 7;
#if USE_ASYNC_LDS
      __builtin_amdgcn_global_load_async_to_lds_b128(
          AS_G1(&Kbase[(size_t)(kt + r) * E_DIM + c8 * 8]),
          AS_L3(&Ks[buf][r * 72 + c8 * 8]), 0, 0);
#else
      *(uint4*)&Ks[buf][r * 72 + c8 * 8] =
          *(const uint4*)&Kbase[(size_t)(kt + r) * E_DIM + c8 * 8];
#endif
    }
  };
  auto loadV = [&](int kt, uint4 vr[2]) {
#pragma unroll
    for (int i = 0; i < 2; ++i) {
      int s = tid + i * 128;
      int kv = s >> 3, d8 = s & 7;
      vr[i] = *(const uint4*)&Vbase[(size_t)(kt + kv) * E_DIM + d8 * 8];
    }
  };
  auto scatterV = [&](int buf, const uint4 vr[2]) {  // transpose into LDS
#pragma unroll
    for (int i = 0; i < 2; ++i) {
      int s = tid + i * 128;
      int kv = s >> 3, d8 = s & 7;
      const __bf16* e = (const __bf16*)&vr[i];
#pragma unroll
      for (int j = 0; j < 8; ++j) Vt[buf][(d8 * 8 + j) * 40 + kv] = e[j];
    }
  };

  // prologue: stage tile 0 into buffer 0
  stageK(0, 0);
  {
    uint4 v0[2];
    loadV(0, v0);
    scatterV(0, v0);
  }

  float mrow[8], lrow[8];
  v8f o[4] = {};
#pragma unroll
  for (int r = 0; r < 8; ++r) { mrow[r] = -3.0e38f; lrow[r] = 0.0f; }

  int cur = 0;
#pragma unroll 1
  for (int kt = 0; kt < S_LEN; kt += 32, cur ^= 1) {
#if USE_ASYNC_LDS
    __builtin_amdgcn_s_wait_asynccnt(0);  // drain this wave's async K writes
#endif
    __syncthreads();  // buf[cur] fully staged by all waves

    const bool hasNext = (kt + 32) < S_LEN;
    uint4 vnext[2];
#if !USE_ASYNC_LDS
    uint4 knext[2];
#endif
    if (hasNext) {
#if USE_ASYNC_LDS
      stageK(cur ^ 1, kt + 32);  // async: fire and forget until next wait
#else
#pragma unroll
      for (int i = 0; i < 2; ++i) {
        int s = tid + i * 128;
        int r = s >> 3, c8 = s & 7;
        knext[i] =
            *(const uint4*)&Kbase[(size_t)(kt + 32 + r) * E_DIM + c8 * 8];
      }
#endif
      loadV(kt + 32, vnext);
    }

    // scores: S[16 x 32] = Q(16x64) @ K_tile^T  (two N tiles, two K slabs)
    v8f s0 = {}, s1 = {};
#pragma unroll
    for (int kq = 0; kq < 2; ++kq) {
      v16bf b0 = frag_colB(&Ks[cur][lmod * 72 + kq * 32 + khalf * 16]);
      v16bf b1 = frag_colB(&Ks[cur][(16 + lmod) * 72 + kq * 32 + khalf * 16]);
      s0 = wmma_bf16(qf[kq], b0, s0);
      s1 = wmma_bf16(qf[kq], b1, s1);
    }

    // scale, mask, online softmax; emit P to per-wave LDS in A-op layout.
#pragma unroll
    for (int r = 0; r < 8; ++r) {
      int qr = qBase + r + 8 * khalf;
      const int* mp = mask + ((size_t)b * S_LEN + qr) * S_LEN + kt;
      float a0 = s0[r] * 0.125f;  // 1/sqrt(64)
      float a1 = s1[r] * 0.125f;
      if (mp[lmod] == 0)      a0 = -1.0e9f;
      if (mp[16 + lmod] == 0) a1 = -1.0e9f;
      float mx = fmaxf(a0, a1);
      mx = fmaxf(mx, __shfl_xor(mx, 1));
      mx = fmaxf(mx, __shfl_xor(mx, 2));
      mx = fmaxf(mx, __shfl_xor(mx, 4));
      mx = fmaxf(mx, __shfl_xor(mx, 8));
      float mnew  = fmaxf(mrow[r], mx);
      float alpha = __expf(mrow[r] - mnew);
      float p0 = __expf(a0 - mnew);
      float p1 = __expf(a1 - mnew);
      float ps = p0 + p1;
      ps += __shfl_xor(ps, 1);
      ps += __shfl_xor(ps, 2);
      ps += __shfl_xor(ps, 4);
      ps += __shfl_xor(ps, 8);
      lrow[r] = lrow[r] * alpha + ps;
      mrow[r] = mnew;
#pragma unroll
      for (int t = 0; t < 4; ++t) o[t][r] *= alpha;
      __bf16* prow = Pw + (r + 8 * khalf) * 40;
      prow[lmod]      = (__bf16)p0;
      prow[16 + lmod] = (__bf16)p1;
    }

    // P buffer is strictly per-wave and same-wave LDS ops are in-order
    // (ISA 7.3), so no workgroup barrier is needed before re-reading it.
    v16bf pf = frag_rowA(Pw + lmod * 40, khalf);
#pragma unroll
    for (int t = 0; t < 4; ++t) {  // O(16x64) += P(16x32) @ V(32x64)
      v16bf vf = frag_colB(&Vt[cur][(t * 16 + lmod) * 40 + khalf * 16]);
      o[t] = wmma_bf16(pf, vf, o[t]);
    }

    if (hasNext) {  // publish next tile (read next iter, after the barrier)
      scatterV(cur ^ 1, vnext);
#if !USE_ASYNC_LDS
#pragma unroll
      for (int i = 0; i < 2; ++i) {
        int s = tid + i * 128;
        int r = s >> 3, c8 = s & 7;
        *(uint4*)&Ks[cur ^ 1][r * 72 + c8 * 8] = knext[i];
      }
#endif
    }
  }

#pragma unroll
  for (int r = 0; r < 8; ++r) {
    float inv = 1.0f / lrow[r];
    int qr = qBase + r + 8 * khalf;
    __bf16* crow = Ctx + ((size_t)b * S_LEN + qr) * E_DIM + h * D_HEAD;
#pragma unroll
    for (int t = 0; t < 4; ++t)
      crow[t * 16 + lmod] = (__bf16)(o[t][r] * inv);
  }
}

extern "C" void kernel_launch(void* const* d_in, const int* in_sizes, int n_in,
                              void* d_out, int out_size, void* d_ws,
                              size_t ws_size, hipStream_t stream) {
  (void)in_sizes; (void)n_in; (void)out_size; (void)ws_size;
  const float* query = (const float*)d_in[0];
  const float* key   = (const float*)d_in[1];
  const float* value = (const float*)d_in[2];
  const int*   maskp = (const int*)d_in[3];
  const float* Wq = (const float*)d_in[4];
  const float* bq = (const float*)d_in[5];
  const float* Wk = (const float*)d_in[6];
  const float* bk = (const float*)d_in[7];
  const float* Wv = (const float*)d_in[8];
  const float* bv = (const float*)d_in[9];
  const float* Wo = (const float*)d_in[10];
  const float* bo = (const float*)d_in[11];

  const size_t nElem = (size_t)M_ROWS * E_DIM;
  __bf16* Qg  = (__bf16*)d_ws;     // 8 MB each, 32 MB total
  __bf16* Kg  = Qg + nElem;
  __bf16* Vg  = Kg + nElem;
  __bf16* Ctx = Vg + nElem;

  dim3 gBlk(256);
  dim3 gGrid(E_DIM / 128, M_ROWS / 128);  // (8, 32)

  gemm_bias_wmma<false, true><<<gGrid, gBlk, 0, stream>>>(
      query, Wq, bq, Qg, M_ROWS, E_DIM, E_DIM);
  gemm_bias_wmma<false, true><<<gGrid, gBlk, 0, stream>>>(
      key, Wk, bk, Kg, M_ROWS, E_DIM, E_DIM);
  gemm_bias_wmma<false, true><<<gGrid, gBlk, 0, stream>>>(
      value, Wv, bv, Vg, M_ROWS, E_DIM, E_DIM);

  flash_attn_wmma<<<dim3(S_LEN / 64, NHEAD, 2), dim3(128), 0, stream>>>(
      Qg, Kg, Vg, maskp, Ctx);

  gemm_bias_wmma<true, false><<<gGrid, gBlk, 0, stream>>>(
      Ctx, Wo, bo, (float*)d_out, M_ROWS, E_DIM, E_DIM);
}
